// TernaryMVAdapter_38886633898334
// MI455X (gfx1250) — 8-bit matrix operand order repaired and checked by reading, compiled, not run on hardware
//
#include <hip/hip_runtime.h>
#include <hip/hip_bf16.h>
#include <cstdint>
#include <cstddef>

#define D_   1152
#define I_   4608
#define L_   256
#define BN_  8
#define NV_  4

typedef int v8i __attribute__((ext_vector_type(8)));
typedef int v4i_t __attribute__((ext_vector_type(4)));

#if defined(__has_builtin)
#if __has_builtin(__builtin_amdgcn_global_load_async_to_lds_b128) && __has_builtin(__builtin_amdgcn_s_wait_asynccnt)
#define ASYNC_COPY 1
#endif
#endif
#ifndef ASYNC_COPY
#define ASYNC_COPY 0
#endif

#if ASYNC_COPY
#define GLD_ASYNC_B128(g, l) \
  __builtin_amdgcn_global_load_async_to_lds_b128( \
      (__attribute__((address_space(1))) v4i_t*)(void*)(g), \
      (__attribute__((address_space(3))) v4i_t*)(void*)(l), 0, 0)
#endif

__device__ __forceinline__ float sigm_(float x){ return 1.f/(1.f+expf(-x)); }

__device__ __forceinline__ float blk_sum(float v, float* sm){
  int t = threadIdx.x;
  sm[t] = v; __syncthreads();
  #pragma unroll
  for (int s = 128; s >= 1; s >>= 1){
    if (t < s) sm[t] += sm[t+s];
    __syncthreads();
  }
  float r = sm[0]; __syncthreads();
  return r;
}
__device__ __forceinline__ float blk_max(float v, float* sm){
  int t = threadIdx.x;
  sm[t] = v; __syncthreads();
  #pragma unroll
  for (int s = 128; s >= 1; s >>= 1){
    if (t < s) sm[t] = fmaxf(sm[t], sm[t+s]);
    __syncthreads();
  }
  float r = sm[0]; __syncthreads();
  return r;
}

__global__ __launch_bounds__(256) void wq_part_kernel(const float* __restrict__ w, long n, float* __restrict__ part){
  __shared__ float sm[256];
  float s = 0.f;
  for (long i = (long)blockIdx.x*256 + threadIdx.x; i < n; i += 256L*256L) s += fabsf(w[i]);
  s = blk_sum(s, sm);
  if (threadIdx.x == 0) part[blockIdx.x] = s;
}
__global__ __launch_bounds__(256) void wq_fin_kernel(const float* __restrict__ part, long n, float* __restrict__ scale){
  __shared__ float sm[256];
  float s = blk_sum(part[threadIdx.x], sm);
  if (threadIdx.x == 0) scale[0] = fmaxf(s/(float)n, 1e-5f);
}
__global__ __launch_bounds__(256) void wq_quant_kernel(const float* __restrict__ w, int8_t* __restrict__ q, long n,
                                                       const float* __restrict__ scale){
  float inv = 1.f/scale[0];
  for (long i = (long)blockIdx.x*256 + threadIdx.x; i < n; i += (long)gridDim.x*256){
    float v = rintf(w[i]*inv);
    q[i] = (int8_t)fminf(fmaxf(v, -1.f), 1.f);
  }
}

__global__ __launch_bounds__(256) void rmsq_kernel(const float* __restrict__ x, int8_t* __restrict__ q,
                                                   float* __restrict__ inv_s, int K){
  __shared__ float sm[256];
  int m = blockIdx.x;
  const float* xr = x + (size_t)m*K;
  float ss = 0.f, mx = 0.f;
  for (int i = threadIdx.x; i < K; i += 256){ float v = xr[i]; ss += v*v; mx = fmaxf(mx, fabsf(v)); }
  ss = blk_sum(ss, sm);
  mx = blk_max(mx, sm);
  float rms = rsqrtf(ss/(float)K + 1e-8f);
  float an  = fmaxf(mx*rms, 1e-5f);
  float sc  = 127.f/an;
  if (threadIdx.x == 0) inv_s[m] = an/127.f;
  int8_t* qr = q + (size_t)m*K;
  for (int i = threadIdx.x; i < K; i += 256){
    float v = rintf(xr[i]*rms*sc);
    qr[i] = (int8_t)fminf(fmaxf(v, -128.f), 127.f);
  }
}

template<int MODE>
__global__ __launch_bounds__(256) void ln_rmsq_kernel(const float* __restrict__ x,
    const float* __restrict__ p0, const float* __restrict__ p1, int mstride, int Lrow, float eps,
    int8_t* __restrict__ q, float* __restrict__ inv_s){
  __shared__ float u[D_];
  __shared__ float sm[256];
  int m = blockIdx.x;
  const float* xr = x + (size_t)m*D_;
  float s1 = 0.f, s2 = 0.f;
  for (int i = threadIdx.x; i < D_; i += 256){ float v = xr[i]; s1 += v; s2 += v*v; }
  s1 = blk_sum(s1, sm);
  s2 = blk_sum(s2, sm);
  float mu = s1/(float)D_;
  float var = s2/(float)D_ - mu*mu;
  float rstd = rsqrtf(var + eps);
  const float* a = p0; const float* b = p1;
  if (MODE == 1){ int bi = m / Lrow; a = p0 + (size_t)bi*mstride; b = p1 + (size_t)bi*mstride; }
  float ss = 0.f, mx = 0.f;
  for (int i = threadIdx.x; i < D_; i += 256){
    float yv = (xr[i]-mu)*rstd;
    float uu = (MODE == 1) ? yv*(1.f + b[i]) + a[i] : yv*a[i] + b[i];
    u[i] = uu; ss += uu*uu; mx = fmaxf(mx, fabsf(uu));
  }
  __syncthreads();
  ss = blk_sum(ss, sm);
  mx = blk_max(mx, sm);
  float rms = rsqrtf(ss/(float)D_ + 1e-8f);
  float an  = fmaxf(mx*rms, 1e-5f);
  float sc  = 127.f/an;
  if (threadIdx.x == 0) inv_s[m] = an/127.f;
  int8_t* qr = q + (size_t)m*D_;
  for (int i = threadIdx.x; i < D_; i += 256){
    float v = rintf(u[i]*rms*sc);
    qr[i] = (int8_t)fminf(fmaxf(v, -128.f), 127.f);
  }
}

__global__ __launch_bounds__(256) void hgrn_outq_kernel(const float* __restrict__ h, const float* __restrict__ g,
                                                        int8_t* __restrict__ q, float* __restrict__ inv_s){
  __shared__ float u[D_];
  __shared__ float sm[256];
  int m = blockIdx.x;
  const float* hr = h + (size_t)m*D_;
  const float* gr = g + (size_t)m*D_;
  float ssh = 0.f;
  for (int i = threadIdx.x; i < D_; i += 256){ float v = hr[i]; ssh += v*v; }
  ssh = blk_sum(ssh, sm);
  float rmsh = rsqrtf(ssh/(float)D_ + 1e-8f);
  float ss = 0.f, mx = 0.f;
  for (int i = threadIdx.x; i < D_; i += 256){
    float gv = gr[i];
    float uu = hr[i]*rmsh*gv*sigm_(gv);
    u[i] = uu; ss += uu*uu; mx = fmaxf(mx, fabsf(uu));
  }
  __syncthreads();
  ss = blk_sum(ss, sm);
  mx = blk_max(mx, sm);
  float rms = rsqrtf(ss/(float)D_ + 1e-8f);
  float an  = fmaxf(mx*rms, 1e-5f);
  float sc  = 127.f/an;
  if (threadIdx.x == 0) inv_s[m] = an/127.f;
  int8_t* qr = q + (size_t)m*D_;
  for (int i = threadIdx.x; i < D_; i += 256){
    float v = rintf(u[i]*rms*sc);
    qr[i] = (int8_t)fminf(fmaxf(v, -128.f), 127.f);
  }
}

__device__ __forceinline__ v8i load_afrag(const int8_t* ap){
  v8i a;
  int4 lo = *(const int4*)(ap);
  int4 hi = *(const int4*)(ap+32);
  a[0] = lo.x; a[1] = lo.y; a[2] = lo.z; a[3] = lo.w;
  a[4] = hi.x; a[5] = hi.y; a[6] = hi.z; a[7] = hi.w;
  return a;
}
__device__ __forceinline__ v8i load_bfrag(const int8_t* bp){
  int4 lo = *(const int4*)(bp);
  int4 hi = *(const int4*)(bp+32);
  v8i b = {lo.x, lo.y, lo.z, lo.w, hi.x, hi.y, hi.z, hi.w};
  return b;
}

template<int EPI>
__global__ __launch_bounds__(256) void gemm_i8(const int8_t* __restrict__ A, const int8_t* __restrict__ W,
    const float* __restrict__ inv_s, const float* __restrict__ wscale, const float* __restrict__ bias,
    float* __restrict__ out, int M, int N, int K){
  const int tid = threadIdx.x, lane = tid & 31, wv = tid >> 5;
  const int m0 = blockIdx.y*128, n0 = blockIdx.x*64;
  const int wr = wv & 3, wc = wv >> 2;
  const int half = lane >> 4, l16 = lane & 15;
  const int sar = tid >> 2;
  const int sac = (tid & 3) << 4;
  const int am0 = min(m0 + sar,      M-1);
  const int am1 = min(m0 + sar + 64, M-1);
  const int wn  = min(n0 + sar,      N-1);
  v8i acc[2][2];
  #pragma unroll
  for (int i = 0; i < 2; ++i)
    #pragma unroll
    for (int j = 0; j < 2; ++j) acc[i][j] = (v8i){0,0,0,0,0,0,0,0};
  const int nk = K >> 6;

#if ASYNC_COPY
  __shared__ __align__(16) int8_t As[2][128*64];
  __shared__ __align__(16) int8_t Ws[2][64*64];
  GLD_ASYNC_B128(A + (size_t)am0*K + sac, &As[0][ sar      *64 + sac]);
  GLD_ASYNC_B128(A + (size_t)am1*K + sac, &As[0][(sar + 64)*64 + sac]);
  GLD_ASYNC_B128(W + (size_t)wn *K + sac, &Ws[0][ sar      *64 + sac]);
  int buf = 0;
  for (int kk = 0; kk < nk; ++kk){
    if (kk + 1 < nk){
      const int k1 = (kk + 1) << 6;
      GLD_ASYNC_B128(A + (size_t)am0*K + k1 + sac, &As[buf^1][ sar      *64 + sac]);
      GLD_ASYNC_B128(A + (size_t)am1*K + k1 + sac, &As[buf^1][(sar + 64)*64 + sac]);
      GLD_ASYNC_B128(W + (size_t)wn *K + k1 + sac, &Ws[buf^1][ sar      *64 + sac]);
      __builtin_amdgcn_s_wait_asynccnt(3);
    } else {
      __builtin_amdgcn_s_wait_asynccnt(0);
    }
    __syncthreads();
    const int8_t* asb = As[buf];
    const int8_t* wsb = Ws[buf];
    v8i a0 = load_afrag(asb + ((wr*32      + l16) << 6) + (half << 4));
    v8i a1 = load_afrag(asb + ((wr*32 + 16 + l16) << 6) + (half << 4));
    #pragma unroll
    for (int fb = 0; fb < 2; ++fb){
      v8i b = load_bfrag(wsb + ((wc*32 + fb*16 + l16) << 6) + (half << 4));
      acc[0][fb] = __builtin_amdgcn_wmma_i32_16x16x64_iu8(true, a0, true, b, acc[0][fb], false, false);
      acc[1][fb] = __builtin_amdgcn_wmma_i32_16x16x64_iu8(true, a1, true, b, acc[1][fb], false, false);
    }
    __syncthreads();
    buf ^= 1;
  }
#else
  __shared__ __align__(16) int8_t As1[128*64];
  __shared__ __align__(16) int8_t Ws1[64*64];
  for (int kk = 0; kk < nk; ++kk){
    const int k0 = kk << 6;
    int4 a0 = *(const int4*)(A + (size_t)am0*K + k0 + sac);
    int4 a1 = *(const int4*)(A + (size_t)am1*K + k0 + sac);
    int4 w4 = *(const int4*)(W + (size_t)wn *K + k0 + sac);
    if (kk + 2 < nk){
      __builtin_prefetch(A + (size_t)am0*K + k0 + 128 + sac, 0, 1);
      __builtin_prefetch(W + (size_t)wn *K + k0 + 128 + sac, 0, 1);
    }
    __syncthreads();
    *(int4*)(As1 + sar*64 + sac)      = a0;
    *(int4*)(As1 + (sar+64)*64 + sac) = a1;
    *(int4*)(Ws1 + sar*64 + sac)      = w4;
    __syncthreads();
    v8i fa0 = load_afrag(As1 + ((wr*32      + l16) << 6) + (half << 4));
    v8i fa1 = load_afrag(As1 + ((wr*32 + 16 + l16) << 6) + (half << 4));
    #pragma unroll
    for (int fb = 0; fb < 2; ++fb){
      v8i b = load_bfrag(Ws1 + ((wc*32 + fb*16 + l16) << 6) + (half << 4));
      acc[0][fb] = __builtin_amdgcn_wmma_i32_16x16x64_iu8(true, fa0, true, b, acc[0][fb], false, false);
      acc[1][fb] = __builtin_amdgcn_wmma_i32_16x16x64_iu8(true, fa1, true, b, acc[1][fb], false, false);
    }
  }
#endif
  const float wsc = wscale[0];
  #pragma unroll
  for (int fa = 0; fa < 2; ++fa){
    const int mb = m0 + wr*32 + fa*16 + half*8;
    #pragma unroll
    for (int j = 0; j < 8; ++j){
      int m = mb + j;
      if (m >= M) continue;
      float s = inv_s[m]*wsc;
      #pragma unroll
      for (int fb = 0; fb < 2; ++fb){
        int n = n0 + wc*32 + fb*16 + l16;
        if (n >= N) continue;
        float y = (float)acc[fa][fb][j]*s;
        if (EPI & 1) y += bias[n];
        if (EPI & 2) y = y*sigm_(y);
        size_t o = (size_t)m*N + n;
        if (EPI & 4) out[o] += y; else out[o] = y;
      }
    }
  }
}

__global__ __launch_bounds__(256) void gemm_i8_swiglu(const int8_t* __restrict__ A, const int8_t* __restrict__ W,
    const float* __restrict__ inv_s, const float* __restrict__ wscale,
    float* __restrict__ out, int M, int NI, int K){
  const int tid = threadIdx.x, lane = tid & 31, wv = tid >> 5;
  const int m0 = blockIdx.y*64, n0 = blockIdx.x*64;
  const int wm = wv & 3, wc = wv >> 2;
  const int half = lane >> 4, l16 = lane & 15;
  const int sar = tid >> 2, sac = (tid & 3) << 4;
  const int am = min(m0 + sar, M-1);
  const int wn = n0 + sar;
  v8i ag0 = {0,0,0,0,0,0,0,0}, ag1 = ag0, au0 = ag0, au1 = ag0;
  const int nk = K >> 6;

#if ASYNC_COPY
  __shared__ __align__(16) int8_t As[2][64*64];
  __shared__ __align__(16) int8_t Wg[2][64*64];
  __shared__ __align__(16) int8_t Wu[2][64*64];
  GLD_ASYNC_B128(A + (size_t)am*K + sac,        &As[0][sar*64 + sac]);
  GLD_ASYNC_B128(W + (size_t)wn*K + sac,        &Wg[0][sar*64 + sac]);
  GLD_ASYNC_B128(W + (size_t)(NI+wn)*K + sac,   &Wu[0][sar*64 + sac]);
  int buf = 0;
  for (int kk = 0; kk < nk; ++kk){
    if (kk + 1 < nk){
      const int k1 = (kk + 1) << 6;
      GLD_ASYNC_B128(A + (size_t)am*K + k1 + sac,      &As[buf^1][sar*64 + sac]);
      GLD_ASYNC_B128(W + (size_t)wn*K + k1 + sac,      &Wg[buf^1][sar*64 + sac]);
      GLD_ASYNC_B128(W + (size_t)(NI+wn)*K + k1 + sac, &Wu[buf^1][sar*64 + sac]);
      __builtin_amdgcn_s_wait_asynccnt(3);
    } else {
      __builtin_amdgcn_s_wait_asynccnt(0);
    }
    __syncthreads();
    v8i av = load_afrag(As[buf] + ((wm*16 + l16) << 6) + (half << 4));
    const int off0 = ((wc*32 + l16) << 6) + (half << 4);
    const int off1 = off0 + (16 << 6);
    v8i g0 = load_bfrag(Wg[buf] + off0), g1 = load_bfrag(Wg[buf] + off1);
    v8i u0 = load_bfrag(Wu[buf] + off0), u1 = load_bfrag(Wu[buf] + off1);
    ag0 = __builtin_amdgcn_wmma_i32_16x16x64_iu8(true, av, true, g0, ag0, false, false);
    ag1 = __builtin_amdgcn_wmma_i32_16x16x64_iu8(true, av, true, g1, ag1, false, false);
    au0 = __builtin_amdgcn_wmma_i32_16x16x64_iu8(true, av, true, u0, au0, false, false);
    au1 = __builtin_amdgcn_wmma_i32_16x16x64_iu8(true, av, true, u1, au1, false, false);
    __syncthreads();
    buf ^= 1;
  }
#else
  __shared__ __align__(16) int8_t As1[64*64];
  __shared__ __align__(16) int8_t Wg1[64*64];
  __shared__ __align__(16) int8_t Wu1[64*64];
  for (int kk = 0; kk < nk; ++kk){
    const int k0 = kk << 6;
    int4 a  = *(const int4*)(A + (size_t)am*K + k0 + sac);
    int4 wg = *(const int4*)(W + (size_t)wn*K + k0 + sac);
    int4 wu = *(const int4*)(W + (size_t)(NI + wn)*K + k0 + sac);
    __syncthreads();
    *(int4*)(As1 + sar*64 + sac) = a;
    *(int4*)(Wg1 + sar*64 + sac) = wg;
    *(int4*)(Wu1 + sar*64 + sac) = wu;
    __syncthreads();
    v8i av = load_afrag(As1 + ((wm*16 + l16) << 6) + (half << 4));
    const int off0 = ((wc*32 + l16) << 6) + (half << 4);
    const int off1 = off0 + (16 << 6);
    v8i g0 = load_bfrag(Wg1 + off0), g1 = load_bfrag(Wg1 + off1);
    v8i u0 = load_bfrag(Wu1 + off0), u1 = load_bfrag(Wu1 + off1);
    ag0 = __builtin_amdgcn_wmma_i32_16x16x64_iu8(true, av, true, g0, ag0, false, false);
    ag1 = __builtin_amdgcn_wmma_i32_16x16x64_iu8(true, av, true, g1, ag1, false, false);
    au0 = __builtin_amdgcn_wmma_i32_16x16x64_iu8(true, av, true, u0, au0, false, false);
    au1 = __builtin_amdgcn_wmma_i32_16x16x64_iu8(true, av, true, u1, au1, false, false);
  }
#endif
  const float wsc = wscale[0];
  const int mb = m0 + wm*16 + half*8;
  const int nb = n0 + wc*32 + l16;
  #pragma unroll
  for (int j = 0; j < 8; ++j){
    int m = mb + j;
    if (m >= M) continue;
    float s = inv_s[m]*wsc;
    {
      float g = (float)ag0[j]*s, u = (float)au0[j]*s;
      out[(size_t)m*NI + nb] = g*sigm_(g)*u;
    }
    {
      float g = (float)ag1[j]*s, u = (float)au1[j]*s;
      out[(size_t)m*NI + nb + 16] = g*sigm_(g)*u;
    }
  }
}

__global__ void hgrn_fv_kernel(float* __restrict__ ib, float* __restrict__ fb, long n){
  long i = (long)blockIdx.x*256 + threadIdx.x;
  if (i >= n) return;
  float f = sigm_(fb[i]);
  float iv = ib[i];
  fb[i] = f;
  ib[i] = iv*sigm_(iv)*(1.f - f);
}
__global__ void hgrn_scan_kernel(const float* __restrict__ f, float* __restrict__ v,
                                 int nseq, int tlen, int inner, int ostride, int tstride){
  int id = blockIdx.x*256 + threadIdx.x;
  if (id >= nseq*D_) return;
  int s = id / D_, d = id - s*D_;
  long base = (long)(s/inner)*ostride + (s % inner);
  float h = 0.f;
  for (int t = 0; t < tlen; ++t){
    size_t off = (size_t)(base + (long)t*tstride)*D_ + d;
    h = fmaf(f[off], h, v[off]);
    v[off] = h;
  }
}

#define PE_TT 8
__global__ __launch_bounds__(256) void patch_embed_kernel(const float* __restrict__ xc, const float* __restrict__ w,
                                                          const float* __restrict__ bias, float* __restrict__ out){
  __shared__ float sp[1536*PE_TT];
  int t0 = blockIdx.x*PE_TT;
  for (int e = threadIdx.x; e < 1536*PE_TT; e += 256){
    int tt = e / 1536, idx = e - tt*1536;
    int token = t0 + tt;
    int n = token >> 10, p = token & 1023;
    int pi = p >> 5, pj = p & 31;
    int c = idx >> 8, r = idx & 255;
    int ki = r >> 4, kj = r & 15;
    sp[idx*PE_TT + tt] = xc[(((size_t)(n*6 + c)*512) + pi*16 + ki)*512 + pj*16 + kj];
  }
  __syncthreads();
  for (int d = threadIdx.x; d < D_; d += 256){
    float acc[PE_TT];
    #pragma unroll
    for (int tt = 0; tt < PE_TT; ++tt) acc[tt] = bias[d];
    const float* wr = w + (size_t)d*1536;
    for (int idx = 0; idx < 1536; ++idx){
      float wv = wr[idx];
      #pragma unroll
      for (int tt = 0; tt < PE_TT; ++tt) acc[tt] = fmaf(wv, sp[idx*PE_TT + tt], acc[tt]);
    }
    for (int tt = 0; tt < PE_TT; ++tt) out[(size_t)(t0+tt)*D_ + d] = acc[tt];
  }
}

__global__ __launch_bounds__(256) void xe_embed_kernel(const float* __restrict__ x, const float* __restrict__ w,
                                                       const float* __restrict__ bias, float* __restrict__ out){
  __shared__ float sp[16];
  int token = blockIdx.x;
  int n = token >> 8, p = token & 255;
  int pi = p >> 4, pj = p & 15;
  if (threadIdx.x < 16){
    int c = threadIdx.x >> 2, r = threadIdx.x & 3;
    int ki = r >> 1, kj = r & 1;
    sp[threadIdx.x] = x[(((size_t)(n*4 + c)*32) + pi*2 + ki)*32 + pj*2 + kj];
  }
  __syncthreads();
  for (int d = threadIdx.x; d < D_; d += 256){
    const float* wr = w + (size_t)d*16;
    float acc = bias[d];
    #pragma unroll
    for (int k = 0; k < 16; ++k) acc = fmaf(wr[k], sp[k], acc);
    out[(size_t)token*D_ + d] = acc;
  }
}

#define DC_TT 2
__global__ __launch_bounds__(256) void downconv_kernel(const float* __restrict__ in, const float* __restrict__ w,
                                                       const float* __restrict__ bias, float* __restrict__ out){
  __shared__ float sp[D_*4*DC_TT];
  int t0 = blockIdx.x*DC_TT;
  for (int e = threadIdx.x; e < D_*4*DC_TT; e += 256){
    int tt = e % DC_TT, cq = e / DC_TT;
    int c = cq >> 2, q = cq & 3;
    int ki = q >> 1, kj = q & 1;
    int token = t0 + tt;
    int n = token >> 8, p = token & 255;
    int oi = p >> 4, oj = p & 15;
    int ii = oi*2 + ki, jj = oj*2 + kj;
    sp[e] = in[(((size_t)n*32 + ii)*32 + jj)*D_ + c];
  }
  __syncthreads();
  for (int d = threadIdx.x; d < D_; d += 256){
    float acc[DC_TT];
    #pragma unroll
    for (int tt = 0; tt < DC_TT; ++tt) acc[tt] = bias[d];
    const float4* wr = (const float4*)(w + (size_t)d*D_*4);
    for (int c = 0; c < D_; ++c){
      float4 wv = wr[c];
      #pragma unroll
      for (int tt = 0; tt < DC_TT; ++tt){
        const float* s = sp + (c*4)*DC_TT + tt;
        acc[tt] = fmaf(wv.x, s[0*DC_TT], acc[tt]);
        acc[tt] = fmaf(wv.y, s[1*DC_TT], acc[tt]);
        acc[tt] = fmaf(wv.z, s[2*DC_TT], acc[tt]);
        acc[tt] = fmaf(wv.w, s[3*DC_TT], acc[tt]);
      }
    }
    for (int tt = 0; tt < DC_TT; ++tt) out[(size_t)(t0+tt)*D_ + d] = acc[tt];
  }
}

__global__ void tfreq_kernel(const float* __restrict__ t, float* __restrict__ emb){
  int m = blockIdx.x, k = threadIdx.x;
  float fr = expf(-logf(10000.f)*(float)k/128.f);
  float a = t[m]*fr;
  emb[m*256 + k]       = cosf(a);
  emb[m*256 + 128 + k] = sinf(a);
}
__global__ void silu_kernel(const float* __restrict__ x, float* __restrict__ y, long n){
  long i = (long)blockIdx.x*256 + threadIdx.x;
  if (i < n){ float v = x[i]; y[i] = v*sigm_(v); }
}
__global__ void addposfeat_kernel(const float* __restrict__ xe, const float* __restrict__ pos,
                                  const float* __restrict__ feat, float* __restrict__ xt, long n){
  long i = (long)blockIdx.x*256 + threadIdx.x;
  if (i >= n) return;
  long m = i / D_; int d = (int)(i - m*D_);
  xt[i] = xe[i] + pos[(m & 255)*D_ + d] + feat[i];
}
__global__ void resid_kernel(float* __restrict__ xt, const float* __restrict__ src,
                             const float* __restrict__ modb, int mstride, int gsel, long n){
  long i = (long)blockIdx.x*256 + threadIdx.x;
  if (i >= n) return;
  long m = i / D_; int d = (int)(i - m*D_);
  int b = (int)(m / L_);
  xt[i] += modb[(size_t)b*mstride + gsel*D_ + d]*src[i];
}
__global__ void refadd_kernel(float* __restrict__ attn, const float* __restrict__ refo, long n){
  long i = (long)blockIdx.x*256 + threadIdx.x;
  if (i >= n) return;
  long m = i / D_; int d = (int)(i - m*D_);
  int bi = (int)(m / (NV_*L_));
  int l  = (int)(m % L_);
  attn[i] += refo[((size_t)bi*L_ + l)*D_ + d];
}
__global__ void unpatch_kernel(const float* __restrict__ fo, float* __restrict__ out){
  int id = blockIdx.x*256 + threadIdx.x;
  if (id >= 65536) return;
  int n = id >> 13, c = (id >> 10) & 7, i2 = (id >> 5) & 31, j2 = id & 31;
  int hp = i2 >> 1, pi = i2 & 1, wp = j2 >> 1, pj = j2 & 1;
  int token = (n*16 + hp)*16 + wp;
  out[id] = fo[(size_t)token*32 + pi*16 + pj*8 + c];
}

extern "C" void kernel_launch(void* const* d_in, const int* in_sizes, int n_in,
                              void* d_out, int out_size, void* d_ws, size_t ws_size,
                              hipStream_t stream){
  (void)in_sizes; (void)n_in; (void)out_size; (void)ws_size;
  const float* X     = (const float*)d_in[0];
  const float* T     = (const float*)d_in[1];
  const float* Y     = (const float*)d_in[2];
  const float* XC    = (const float*)d_in[3];
  const float* REF   = (const float*)d_in[4];
  const float* XE_W  = (const float*)d_in[6];
  const float* XE_B  = (const float*)d_in[7];
  const float* POS   = (const float*)d_in[8];
  const float* TE_W1 = (const float*)d_in[9];
  const float* TE_B1 = (const float*)d_in[10];
  const float* TE_W2 = (const float*)d_in[11];
  const float* TE_B2 = (const float*)d_in[12];
  const float* PP_W  = (const float*)d_in[13];
  const float* PP_B  = (const float*)d_in[14];
  const float* SA_PE_W = (const float*)d_in[15];
  const float* SA_PE_B = (const float*)d_in[16];
  const float* SA_LN_G = (const float*)d_in[17];
  const float* SA_LN_B = (const float*)d_in[18];
  const float* SA_GATE = (const float*)d_in[19];
  const float* SA_DOWN = (const float*)d_in[20];
  const float* SA_DC_W = (const float*)d_in[21];
  const float* SA_DC_B = (const float*)d_in[22];
  const float* ADA_W = (const float*)d_in[23];
  const float* ADA_B = (const float*)d_in[24];
  const float* W_SA[4] = {(const float*)d_in[25], (const float*)d_in[26], (const float*)d_in[27], (const float*)d_in[28]};
  const float* W_MV[4] = {(const float*)d_in[29], (const float*)d_in[30], (const float*)d_in[31], (const float*)d_in[32]};
  const float* W_RF[4] = {(const float*)d_in[33], (const float*)d_in[34], (const float*)d_in[35], (const float*)d_in[36]};
  const float* GATE_W = (const float*)d_in[37];
  const float* DOWN_W = (const float*)d_in[38];
  const float* FL_W  = (const float*)d_in[39];
  const float* FL_B  = (const float*)d_in[40];
  const float* FLA_W = (const float*)d_in[41];
  const float* FLA_B = (const float*)d_in[42];

  char* base = (char*)d_ws;
  size_t off = 0;
  auto alloc = [&](size_t bytes)->char*{
    char* p = base + off;
    off = (off + bytes + 255) & ~(size_t)255;
    return p;
  };

  float* wscales  = (float*)alloc(64*sizeof(float));
  float* partials = (float*)alloc(256*sizeof(float));

  const float* Wsrc[40]; size_t Wn[40]; int8_t* Wq[40]; int nW = 0;
  auto addw = [&](const float* p, size_t elems)->int{
    Wsrc[nW] = p; Wn[nW] = elems; Wq[nW] = (int8_t*)alloc(elems);
    return nW++;
  };
  int q_te1 = addw(TE_W1, (size_t)1152*256);
  int q_te2 = addw(TE_W2, (size_t)1152*1152);
  int q_pp  = addw(PP_W,  (size_t)1152*768);
  int q_sagate = addw(SA_GATE, (size_t)9216*1152);
  int q_sadown = addw(SA_DOWN, (size_t)1152*4608);
  int q_ada[2], q_sa[2][4], q_mv[2][4], q_rf[2][4], q_gate[2], q_down[2];
  for (int l = 0; l < 2; ++l){
    q_ada[l] = addw(ADA_W + (size_t)l*6912*1152, (size_t)6912*1152);
    for (int k = 0; k < 4; ++k){
      q_sa[l][k] = addw(W_SA[k] + (size_t)l*1152*1152, (size_t)1152*1152);
      q_mv[l][k] = addw(W_MV[k] + (size_t)l*1152*1152, (size_t)1152*1152);
      q_rf[l][k] = addw(W_RF[k] + (size_t)l*1152*1152, (size_t)1152*1152);
    }
    q_gate[l] = addw(GATE_W + (size_t)l*9216*1152, (size_t)9216*1152);
    q_down[l] = addw(DOWN_W + (size_t)l*1152*4608, (size_t)1152*4608);
  }
  int q_flada = addw(FLA_W, (size_t)2304*1152);
  int q_fl    = addw(FL_W,  (size_t)32*1152);

  float*  invA = (float*)alloc(8192*sizeof(float));
  float*  invB = (float*)alloc(2048*sizeof(float));
  float*  invC = (float*)alloc(16*sizeof(float));
  int8_t* a8a  = (int8_t*)alloc((size_t)8192*4608);
  int8_t* a8b  = (int8_t*)alloc((size_t)2048*1152);
  int8_t* a8r  = (int8_t*)alloc((size_t)512*1152);
  int8_t* a8c  = (int8_t*)alloc((size_t)16*1152);
  float*  invR = (float*)alloc(512*sizeof(float));
  float* htok  = (float*)alloc((size_t)8192*1152*4);
  float* gated = (float*)alloc((size_t)8192*4608*4);
  float* mlpo  = (float*)alloc((size_t)8192*1152*4);
  float* feat0 = (float*)alloc((size_t)2048*1152*4);
  float* xt    = (float*)alloc((size_t)2048*1152*4);
  float* ibuf  = (float*)alloc((size_t)2048*1152*4);
  float* fbuf  = (float*)alloc((size_t)2048*1152*4);
  float* gbuf  = (float*)alloc((size_t)2048*1152*4);
  float* attns = (float*)alloc((size_t)2048*1152*4);
  float* refo  = (float*)alloc((size_t)512*1152*4);
  float* tfre  = (float*)alloc((size_t)16*256*4);
  float* teh   = (float*)alloc((size_t)16*1152*4);
  float* cbuf  = (float*)alloc((size_t)16*1152*4);
  float* csil  = (float*)alloc((size_t)16*1152*4);
  float* modb  = (float*)alloc((size_t)16*6912*4);
  float* modfl = (float*)alloc((size_t)16*2304*4);
  float* flo   = (float*)alloc((size_t)2048*32*4);

  for (int i = 0; i < nW; ++i){
    wq_part_kernel<<<256, 256, 0, stream>>>(Wsrc[i], (long)Wn[i], partials);
    wq_fin_kernel<<<1, 256, 0, stream>>>(partials, (long)Wn[i], wscales + i);
    wq_quant_kernel<<<1024, 256, 0, stream>>>(Wsrc[i], Wq[i], (long)Wn[i], wscales + i);
  }

  auto G = [&](const int8_t* A, int qw, const float* inv, const float* bias,
               float* outp, int M, int N, int K, int flags){
    dim3 gr((N + 63)/64, (M + 127)/128);
    const int8_t* Wp = Wq[qw]; const float* sc = wscales + qw;
    switch (flags){
      case 1: gemm_i8<1><<<gr, 256, 0, stream>>>(A, Wp, inv, sc, bias, outp, M, N, K); break;
      case 3: gemm_i8<3><<<gr, 256, 0, stream>>>(A, Wp, inv, sc, bias, outp, M, N, K); break;
      case 4: gemm_i8<4><<<gr, 256, 0, stream>>>(A, Wp, inv, sc, bias, outp, M, N, K); break;
      case 5: gemm_i8<5><<<gr, 256, 0, stream>>>(A, Wp, inv, sc, bias, outp, M, N, K); break;
      default: gemm_i8<0><<<gr, 256, 0, stream>>>(A, Wp, inv, sc, bias, outp, M, N, K); break;
    }
  };
  auto SW = [&](const int8_t* A, int qw, const float* inv, float* outp, int M){
    dim3 gr(I_/64, (M + 63)/64);
    gemm_i8_swiglu<<<gr, 256, 0, stream>>>(A, Wq[qw], inv, wscales + qw, outp, M, I_, D_);
  };
  auto HG = [&](const int8_t* A8, const float* inv, int M, const int* qw,
                int nseq, int tlen, int inner, int ostr, int tstr, float* outb, int accum){
    G(A8, qw[0], inv, nullptr, ibuf, M, D_, D_, 0);
    G(A8, qw[1], inv, nullptr, fbuf, M, D_, D_, 0);
    G(A8, qw[2], inv, nullptr, gbuf, M, D_, D_, 0);
    long n = (long)M*D_;
    hgrn_fv_kernel<<<(int)((n + 255)/256), 256, 0, stream>>>(ibuf, fbuf, n);
    int th = nseq*D_;
    hgrn_scan_kernel<<<(th + 255)/256, 256, 0, stream>>>(fbuf, ibuf, nseq, tlen, inner, ostr, tstr);
    hgrn_outq_kernel<<<M, 256, 0, stream>>>(ibuf, gbuf, a8b, invB);
    G(a8b, qw[3], invB, nullptr, outb, M, D_, D_, accum ? 4 : 0);
  };

  tfreq_kernel<<<8, 128, 0, stream>>>(T, tfre);
  rmsq_kernel<<<8, 256, 0, stream>>>(tfre, a8c, invC, 256);
  G(a8c, q_te1, invC, TE_B1, teh, 8, 1152, 256, 3);
  rmsq_kernel<<<8, 256, 0, stream>>>(teh, a8c, invC, 1152);
  G(a8c, q_te2, invC, TE_B2, cbuf, 8, 1152, 1152, 1);
  rmsq_kernel<<<8, 256, 0, stream>>>(Y, a8c, invC, 768);
  G(a8c, q_pp, invC, PP_B, cbuf, 8, 1152, 768, 5);
  silu_kernel<<<(8*1152 + 255)/256, 256, 0, stream>>>(cbuf, csil, 8L*1152);
  rmsq_kernel<<<8, 256, 0, stream>>>(csil, a8c, invC, 1152);

  patch_embed_kernel<<<8192/PE_TT, 256, 0, stream>>>(XC, SA_PE_W, SA_PE_B, htok);
  ln_rmsq_kernel<0><<<8192, 256, 0, stream>>>(htok, SA_LN_G, SA_LN_B, 0, 1, 1e-5f, a8a, invA);
  SW(a8a, q_sagate, invA, gated, 8192);
  rmsq_kernel<<<8192, 256, 0, stream>>>(gated, a8a, invA, 4608);
  G(a8a, q_sadown, invA, nullptr, mlpo, 8192, 1152, 4608, 0);
  downconv_kernel<<<2048/DC_TT, 256, 0, stream>>>(mlpo, SA_DC_W, SA_DC_B, feat0);

  xe_embed_kernel<<<2048, 256, 0, stream>>>(X, XE_W, XE_B, ibuf);
  addposfeat_kernel<<<(int)((2048L*1152 + 255)/256), 256, 0, stream>>>(ibuf, POS, feat0, xt, 2048L*1152);

  rmsq_kernel<<<512, 256, 0, stream>>>(REF, a8r, invR, 1152);

  long n2048 = 2048L*1152;
  for (int l = 0; l < 2; ++l){
    G(a8c, q_ada[l], invC, ADA_B + (size_t)l*6912, modb, 8, 6912, 1152, 1);
    ln_rmsq_kernel<1><<<2048, 256, 0, stream>>>(xt, modb + 0, modb + 1152, 6912, 256, 1e-6f, a8a, invA);
    HG(a8a, invA, 2048, q_sa[l], 8,   256, 1,   256,  1,   attns, 0);
    HG(a8a, invA, 2048, q_mv[l], 512, 4,   256, 1024, 256, attns, 1);
    HG(a8r, invR, 512,  q_rf[l], 2,   256, 1,   256,  1,   refo,  0);
    refadd_kernel<<<(int)((n2048 + 255)/256), 256, 0, stream>>>(attns, refo, n2048);
    resid_kernel<<<(int)((n2048 + 255)/256), 256, 0, stream>>>(xt, attns, modb, 6912, 2, n2048);
    ln_rmsq_kernel<1><<<2048, 256, 0, stream>>>(xt, modb + 3*1152, modb + 4*1152, 6912, 256, 1e-6f, a8a, invA);
    SW(a8a, q_gate[l], invA, gated, 2048);
    rmsq_kernel<<<2048, 256, 0, stream>>>(gated, a8a, invA, 4608);
    G(a8a, q_down[l], invA, nullptr, mlpo, 2048, 1152, 4608, 0);
    resid_kernel<<<(int)((n2048 + 255)/256), 256, 0, stream>>>(xt, mlpo, modb, 6912, 5, n2048);
  }

  G(a8c, q_flada, invC, FLA_B, modfl, 8, 2304, 1152, 1);
  ln_rmsq_kernel<1><<<2048, 256, 0, stream>>>(xt, modfl + 0, modfl + 1152, 2304, 256, 1e-6f, a8a, invA);
  G(a8a, q_fl, invA, FL_B, flo, 2048, 32, 1152, 1);
  unpatch_kernel<<<256, 256, 0, stream>>>(flo, (float*)d_out);
}
